// MinCutNet_readout_embedding_32006096289958
// MI455X (gfx1250) — matrix operand order repaired and checked by reading, not compiled, not run on hardware
//
#include <hip/hip_runtime.h>
#include <hip/hip_bf16.h>
#include <math.h>

// MinCutPool net on MI455X (gfx1250, wave32, WMMA bf16).
// One 256-thread workgroup (8 wave32) per graph; dense 256x256 f32 adjacency in LDS
// (needs CDNA5's 320KB/WGP LDS); all O(N^2 H)/O(N FIN H) matmuls on v_wmma_f32_16x16x32_bf16.

typedef __attribute__((ext_vector_type(16))) __bf16 v16bf;
typedef __attribute__((ext_vector_type(8)))  float  v8f;

#define BGRAPH 512
#define NNODE  256
#define EPER   8192
#define FINC   128
#define HDIM   32
#define KCLUS  16
#define FOUTC  10

// ---- LDS layout (bytes) -------------------------------------------------
#define OFF_ADJ   0        // f32  [256][256]  adjacency counts      (262144)
#define OFF_HT    262144   // bf16 [32][256]   h0 transposed  -> reused: h1 row-major bf16 [256][32]
#define OFF_ARM   278528   // bf16 [256][32]   h0 row-major   -> reused: sTA f32 [16][256]
#define OFF_TRM   294912   // bf16 [256][32]   t row-major    -> reused: softmax s f32 [256][16]
#define OFF_POOLX 311296   // f32  [16][32]
#define OFF_T2    313344   // f32  [16][32]
#define OFF_H2    315392   // f32  [16][32]
#define OFF_OADJ  317440   // f32  [16][16]
#define OFF_SS    318464   // f32  [16][16]
#define OFF_DSQ   319488   // f32  [16]
#define OFF_RV    319552   // f32  [32]
#define OFF_QV    319680   // f32  [32]
#define OFF_OV    319808   // f32  [16]
#define OFF_DEN   319872   // f32  [1]
#define SMEM_BYTES 319904  // < 327680 (320KB WGP cap)

// ---- WMMA helpers: C/D per CDNA5 ISA §7.12.2, A/B in contiguous K order ------------------------
__device__ __forceinline__ v8f wmma_bf16(v16bf a, v16bf b, v8f c) {
  return __builtin_amdgcn_wmma_f32_16x16x32_bf16(false, a, false, b, (short)0, c, false, false);
}

// A (16x32 bf16): lane m=L&15, half=L>>4; elems i -> K=half*16+i
__device__ __forceinline__ v16bf a_tile_f32(const float* M, int ld, int m0, int k0) {
  int lane = threadIdx.x & 31;
  const float* p = M + (m0 + (lane & 15)) * ld + k0 + (lane >> 4) * 16;
  v16bf a;
#pragma unroll
  for (int i = 0; i < 8; ++i) a[i] = (__bf16)p[i];
#pragma unroll
  for (int i = 0; i < 8; ++i) a[i + 8] = (__bf16)p[i + 8];
  return a;
}
__device__ __forceinline__ v16bf a_tile_bf16(const __bf16* M, int ld, int m0, int k0) {
  int lane = threadIdx.x & 31;
  const __bf16* p = M + (m0 + (lane & 15)) * ld + k0 + (lane >> 4) * 16;
  v16bf a;
#pragma unroll
  for (int i = 0; i < 8; ++i) a[i] = p[i];
#pragma unroll
  for (int i = 0; i < 8; ++i) a[i + 8] = p[i + 8];
  return a;
}
// B (32x16): lane n=L&15, half=L>>4; elems i -> K=half*16+i. BT is [ncols][ldk] (transposed buffer).
__device__ __forceinline__ v16bf b_tile_bf16T(const __bf16* BT, int ld, int n0, int k0) {
  int lane = threadIdx.x & 31;
  const __bf16* p = BT + (n0 + (lane & 15)) * ld + k0 + (lane >> 4) * 16;
  v16bf b;
#pragma unroll
  for (int i = 0; i < 16; ++i) b[i] = p[i];
  return b;
}
// B from row-major f32 weight W[K][ld] (strided; tiny matrices, L2-resident)
__device__ __forceinline__ v16bf b_tile_f32RM(const float* W, int ld, int n0, int k0) {
  int lane = threadIdx.x & 31;
  int n = n0 + (lane & 15);
  int kb = k0 + (lane >> 4) * 16;
  v16bf b;
#pragma unroll
  for (int i = 0; i < 16; ++i) b[i] = (__bf16)W[(kb + i) * ld + n];
  return b;
}
// C/D: VGPR g, lane L -> N=L&15, M=(L>>4)*8+g
__device__ __forceinline__ v8f bias_acc(const float* bias, int n0) {
  float bv = bias[n0 + (threadIdx.x & 15)];
  v8f c;
#pragma unroll
  for (int g = 0; g < 8; ++g) c[g] = bv;
  return c;
}
__device__ __forceinline__ void store_d_bf16RM(__bf16* O, int ld, int m0, int n0, v8f acc) {
  int lane = threadIdx.x & 31;
  int n = n0 + (lane & 15), mb = m0 + (lane >> 4) * 8;
#pragma unroll
  for (int g = 0; g < 8; ++g) O[(mb + g) * ld + n] = (__bf16)acc[g];
}
__device__ __forceinline__ void store_d_bf16T(__bf16* OT, int ld, int m0, int n0, v8f acc) {
  int lane = threadIdx.x & 31;
  int n = n0 + (lane & 15), mb = m0 + (lane >> 4) * 8;
  __bf16* p = OT + n * ld + mb;  // 8 consecutive M -> contiguous 16B store
#pragma unroll
  for (int g = 0; g < 8; ++g) p[g] = (__bf16)acc[g];
}

__global__ void mincut_zero_losses(float* out) {
  if (threadIdx.x < 2) out[BGRAPH * FOUTC + threadIdx.x] = 0.0f;
}

__global__ void __launch_bounds__(256, 1)
mincut_graph_kernel(const float* __restrict__ x, const long long* __restrict__ ei,
                    const float* __restrict__ W1, const float* __restrict__ b1,
                    const float* __restrict__ Wrel1, const float* __restrict__ brel1,
                    const float* __restrict__ Wroot1,
                    const float* __restrict__ Wpool, const float* __restrict__ bpool,
                    const float* __restrict__ Wrel2, const float* __restrict__ brel2,
                    const float* __restrict__ Wroot2,
                    const float* __restrict__ Wlin2, const float* __restrict__ blin2,
                    const float* __restrict__ Wlin3, const float* __restrict__ blin3,
                    float* __restrict__ out) {
  extern __shared__ char smem[];
  float*  adjf  = (float*)(smem + OFF_ADJ);
  __bf16* hT    = (__bf16*)(smem + OFF_HT);
  __bf16* h1RM  = (__bf16*)(smem + OFF_HT);    // alias (after stage 4)
  __bf16* aRM   = (__bf16*)(smem + OFF_ARM);
  float*  sTA   = (float*)(smem + OFF_ARM);    // alias (after stage 5)
  __bf16* tRM   = (__bf16*)(smem + OFF_TRM);
  float*  sS    = (float*)(smem + OFF_TRM);    // alias (after stage 5)
  float*  poolx = (float*)(smem + OFF_POOLX);
  float*  t2    = (float*)(smem + OFF_T2);
  float*  h2    = (float*)(smem + OFF_H2);
  float*  oadj  = (float*)(smem + OFF_OADJ);
  float*  ssM   = (float*)(smem + OFF_SS);
  float*  dsq   = (float*)(smem + OFF_DSQ);
  float*  rv    = (float*)(smem + OFF_RV);
  float*  qv    = (float*)(smem + OFF_QV);
  float*  ov    = (float*)(smem + OFF_OV);
  float*  denS  = (float*)(smem + OFF_DEN);

  const int b    = blockIdx.x;
  const int tid  = threadIdx.x;
  const int wave = tid >> 5;

  // ---- stage 1: zero adjacency + scalars ----
  for (int i = tid; i < NNODE * NNODE; i += 256) adjf[i] = 0.0f;
  if (tid == 0) denS[0] = 0.0f;
  __syncthreads();

  // ---- stage 2: scatter-build adjacency (LDS f32 atomics) ----
  {
    const long long base = (long long)b * EPER;
    const long long off  = (long long)b * NNODE;
    for (int e = tid; e < EPER; e += 256) {
      int s = (int)(ei[base + e] - off);
      int d = (int)(ei[(long long)BGRAPH * EPER + base + e] - off);
      atomicAdd(&adjf[s * NNODE + d], 1.0f);
    }
  }

  // ---- stage 3: h0 = x @ W1 + b1  (WMMA; A from global f32, B strided from W1) ----
  {
    const float* xg = x + (long long)b * NNODE * FINC;
#pragma unroll
    for (int j = 0; j < 4; ++j) {
      int t = wave + 8 * j;             // 32 tiles over 8 waves
      int mt = (t & 15) * 16, nt = (t >> 4) * 16;
      v8f acc = bias_acc(b1, nt);
      for (int kk = 0; kk < FINC; kk += 32) {
        v16bf a = a_tile_f32(xg, FINC, mt, kk);
        v16bf bm = b_tile_f32RM(W1, HDIM, nt, kk);
        acc = wmma_bf16(a, bm, acc);
      }
      store_d_bf16RM(aRM, HDIM, mt, nt, acc);   // h0 row-major (A-operand later)
      store_d_bf16T(hT, NNODE, mt, nt, acc);    // h0 transposed (B-operand)
    }
  }
  __syncthreads();

  // ---- stage 4: t = adj @ h0  (the big WMMA: K=256) ----
#pragma unroll
  for (int j = 0; j < 4; ++j) {
    int t = wave + 8 * j;
    int mt = (t & 15) * 16, nt = (t >> 4) * 16;
    v8f acc = {};
    for (int kk = 0; kk < NNODE; kk += 32) {
      v16bf a = a_tile_f32(adjf, NNODE, mt, kk);   // counts exact in bf16
      v16bf bm = b_tile_bf16T(hT, NNODE, nt, kk);
      acc = wmma_bf16(a, bm, acc);
    }
    store_d_bf16RM(tRM, HDIM, mt, nt, acc);
  }
  __syncthreads();

  // ---- stage 5: h1 = t @ Wrel1 + h0 @ Wroot1 + brel1 ----
#pragma unroll
  for (int j = 0; j < 4; ++j) {
    int t = wave + 8 * j;
    int mt = (t & 15) * 16, nt = (t >> 4) * 16;
    v8f acc = bias_acc(brel1, nt);
    acc = wmma_bf16(a_tile_bf16(tRM, HDIM, mt, 0), b_tile_f32RM(Wrel1, HDIM, nt, 0), acc);
    acc = wmma_bf16(a_tile_bf16(aRM, HDIM, mt, 0), b_tile_f32RM(Wroot1, HDIM, nt, 0), acc);
    store_d_bf16RM(h1RM, HDIM, mt, nt, acc);  // overwrites hT region (dead)
  }
  __syncthreads();

  // ---- stage 6: s = softmax(h1 @ Wpool + bpool); d_flat; mincut_den ----
  {
    int n = tid;
    float hv[HDIM];
#pragma unroll
    for (int i = 0; i < HDIM; ++i) hv[i] = (float)h1RM[n * HDIM + i];
    float lg[KCLUS];
#pragma unroll
    for (int k = 0; k < KCLUS; ++k) {
      float a = bpool[k];
      for (int h = 0; h < HDIM; ++h) a += hv[h] * Wpool[h * KCLUS + k];
      lg[k] = a;
    }
    float mx = lg[0];
#pragma unroll
    for (int k = 1; k < KCLUS; ++k) mx = fmaxf(mx, lg[k]);
    float se = 0.0f;
#pragma unroll
    for (int k = 0; k < KCLUS; ++k) { lg[k] = __expf(lg[k] - mx); se += lg[k]; }
    float inv = 1.0f / se, ssq = 0.0f;
#pragma unroll
    for (int k = 0; k < KCLUS; ++k) { float v = lg[k] * inv; sS[n * KCLUS + k] = v; ssq += v * v; }
    float dsum = 0.0f;
    for (int mm = 0; mm < NNODE; ++mm) dsum += adjf[n * NNODE + ((mm + tid) & 255)];  // bank-staggered
    atomicAdd(denS, dsum * ssq);
  }
  __syncthreads();

  // ---- stage 7: sTA[k][m] = sum_n s[n][k] * adj[n][m];  ss = s^T s ----
  {
    int m = tid;
    float acc[KCLUS];
#pragma unroll
    for (int k = 0; k < KCLUS; ++k) acc[k] = 0.0f;
    for (int n = 0; n < NNODE; ++n) {
      float a = adjf[n * NNODE + m];
#pragma unroll
      for (int k = 0; k < KCLUS; ++k) acc[k] += sS[n * KCLUS + k] * a;
    }
#pragma unroll
    for (int k = 0; k < KCLUS; ++k) sTA[k * NNODE + m] = acc[k];
    int k = tid >> 4, jj = tid & 15;
    float ss = 0.0f;
    for (int n = 0; n < NNODE; ++n) ss += sS[n * KCLUS + k] * sS[n * KCLUS + jj];
    ssM[tid] = ss;
  }
  __syncthreads();

  // ---- stage 8: out_adj = sTA @ s ; pooled x = s^T h1 ----
  {
    int k = tid >> 4, jj = tid & 15;
    float a = 0.0f;
    for (int m = 0; m < NNODE; ++m) a += sTA[k * NNODE + m] * sS[m * KCLUS + jj];
    oadj[tid] = a;
#pragma unroll
    for (int r = 0; r < 2; ++r) {
      int e = tid + r * 256, kk = e >> 5, h = e & 31;
      float p = 0.0f;
      for (int n = 0; n < NNODE; ++n) p += sS[n * KCLUS + kk] * (float)h1RM[n * HDIM + h];
      poolx[e] = p;
    }
  }
  __syncthreads();

  // ---- stage 9: losses + degree-normalized pooled adjacency ----
  if (tid == 0) {
    float num = 0.0f;
    for (int k = 0; k < KCLUS; ++k) num += oadj[k * KCLUS + k];
    atomicAdd(out + BGRAPH * FOUTC, -(num / denS[0]) * (1.0f / BGRAPH));
    float nrm = 0.0f;
    for (int i = 0; i < KCLUS * KCLUS; ++i) nrm += ssM[i] * ssM[i];
    nrm = sqrtf(nrm);
    float ortho = 0.0f;
    for (int k = 0; k < KCLUS; ++k)
      for (int jj = 0; jj < KCLUS; ++jj) {
        float v = ssM[k * KCLUS + jj] / nrm - (k == jj ? 0.25f : 0.0f);  // I/sqrt(16)
        ortho += v * v;
      }
    atomicAdd(out + BGRAPH * FOUTC + 1, sqrtf(ortho) * (1.0f / BGRAPH));
  }
  if (tid < KCLUS) {
    float rs = 0.0f;
    for (int jj = 0; jj < KCLUS; ++jj) if (jj != tid) rs += oadj[tid * KCLUS + jj];
    dsq[tid] = sqrtf(rs) + 1e-15f;
  }
  __syncthreads();
  {
    int k = tid >> 4, jj = tid & 15;
    oadj[tid] = (k == jj) ? 0.0f : oadj[tid] / (dsq[jj] * dsq[k]);
  }
  __syncthreads();

  // ---- stage 10: conv2 on pooled graph + readout + MLP + log_softmax ----
#pragma unroll
  for (int r = 0; r < 2; ++r) {
    int e = tid + r * 256, k = e >> 5, h = e & 31;
    float a = 0.0f;
#pragma unroll
    for (int jj = 0; jj < KCLUS; ++jj) a += oadj[k * KCLUS + jj] * poolx[jj * HDIM + h];
    t2[e] = a;
  }
  __syncthreads();
#pragma unroll
  for (int r = 0; r < 2; ++r) {
    int e = tid + r * 256, k = e >> 5, h = e & 31;
    float a = brel2[h];
    for (int q = 0; q < HDIM; ++q)
      a += t2[k * HDIM + q] * Wrel2[q * HDIM + h] + poolx[k * HDIM + q] * Wroot2[q * HDIM + h];
    h2[e] = a;
  }
  __syncthreads();
  if (tid < HDIM) {
    float r = 0.0f;
    for (int k = 0; k < KCLUS; ++k) r += h2[k * HDIM + tid];
    rv[tid] = r;
  }
  __syncthreads();
  if (tid < HDIM) {
    float q = blin2[tid];
    for (int h = 0; h < HDIM; ++h) q += rv[h] * Wlin2[h * HDIM + tid];
    qv[tid] = fmaxf(q, 0.0f);
  }
  __syncthreads();
  if (tid < FOUTC) {
    float o = blin3[tid];
    for (int jj = 0; jj < HDIM; ++jj) o += qv[jj] * Wlin3[jj * FOUTC + tid];
    ov[tid] = o;
  }
  __syncthreads();
  if (tid < FOUTC) {
    float mx = ov[0];
    for (int c = 1; c < FOUTC; ++c) mx = fmaxf(mx, ov[c]);
    float se = 0.0f;
    for (int c = 0; c < FOUTC; ++c) se += __expf(ov[c] - mx);
    out[b * FOUTC + tid] = ov[tid] - mx - __logf(se);
  }
}

extern "C" void kernel_launch(void* const* d_in, const int* in_sizes, int n_in,
                              void* d_out, int out_size, void* d_ws, size_t ws_size,
                              hipStream_t stream) {
  (void)in_sizes; (void)n_in; (void)out_size; (void)d_ws; (void)ws_size;
  const float*     x      = (const float*)d_in[0];
  const long long* ei     = (const long long*)d_in[1];
  // d_in[2] = batch (layout is implied: repeat(arange(B), N)) — unused
  const float* W1     = (const float*)d_in[3];
  const float* b1     = (const float*)d_in[4];
  const float* Wrel1  = (const float*)d_in[5];
  const float* brel1  = (const float*)d_in[6];
  const float* Wroot1 = (const float*)d_in[7];
  const float* Wpool  = (const float*)d_in[8];
  const float* bpool  = (const float*)d_in[9];
  const float* Wrel2  = (const float*)d_in[10];
  const float* brel2  = (const float*)d_in[11];
  const float* Wroot2 = (const float*)d_in[12];
  const float* Wlin2  = (const float*)d_in[13];
  const float* blin2  = (const float*)d_in[14];
  const float* Wlin3  = (const float*)d_in[15];
  const float* blin3  = (const float*)d_in[16];
  float* out = (float*)d_out;

  mincut_zero_losses<<<1, 32, 0, stream>>>(out);
  mincut_graph_kernel<<<BGRAPH, 256, SMEM_BYTES, stream>>>(
      x, ei, W1, b1, Wrel1, brel1, Wroot1, Wpool, bpool,
      Wrel2, brel2, Wroot2, Wlin2, blin2, Wlin3, blin3, out);
}
